// TransformerDecoderBlock_7241314861343
// MI455X (gfx1250) — matrix operand order repaired and checked by reading, not compiled, not run on hardware
//
#include <hip/hip_runtime.h>
#include <hip/hip_bf16.h>

// ---------------------------------------------------------------------------
// Transformer decoder block (LoRA + alibi + SwiGLU + layer-scale) for MI455X
// gfx1250. All GEMMs and attention matmuls use v_wmma_f32_16x16x32_bf16.
// fp32 inputs are converted to bf16 for the matrix paths (fp32 accumulate).
// ---------------------------------------------------------------------------

#define D_MODEL 2048
#define N_HEADS 16
#define HEAD_DIM 128
#define D_FF 8192
#define RANK 32
#define SEQ 2048
#define BATCH 2
#define TOKENS (BATCH * SEQ)   // 4096

typedef __bf16 bf16;
typedef __attribute__((ext_vector_type(8)))  __bf16 v8bf;
typedef __attribute__((ext_vector_type(16))) __bf16 v16bf;
typedef __attribute__((ext_vector_type(8)))  float  v8f;

static __device__ inline v16bf ld_frag(const bf16* p0, const bf16* p1) {
    union { v16bf v; v8bf h[2]; } u;
    u.h[0] = *(const v8bf*)p0;
    u.h[1] = *(const v8bf*)p1;
    return u.v;
}

static __device__ inline v8f wmma_bf16(v16bf a, v16bf b, v8f c) {
    // D = A(16x32) * B(32x16) + C, fp32 accumulate
    return __builtin_amdgcn_wmma_f32_16x16x32_bf16(
        false, a, false, b, (short)0, c, false, false);
}

// ---------------------------------------------------------------------------
// Generic bf16 WMMA GEMM:  C[M,N] (f32) (+)= scale * A[M,K] * W[N,K]^T
// A row-major bf16 (lda), W row-major bf16 [N,K] (ldb) => both K-contiguous.
// Block: 256 threads = 8 waves; wave computes a 16 x (16*NT) tile.
// Grid: x = N/(16*NT), y = M/128 (M is always 4096 here).
// NT is a template parameter so the K-loop has NO control flow (no spills).
// ---------------------------------------------------------------------------
template <int NT>
__global__ void tdb_gemm_bf16(const bf16* __restrict__ A, int lda,
                              const bf16* __restrict__ W, int ldb,
                              float* __restrict__ C, int ldc,
                              int K, float scale, int accum) {
    const int lane = threadIdx.x & 31;
    const int wave = threadIdx.x >> 5;
    const int half = lane >> 4;     // 0: lanes 0-15, 1: lanes 16-31
    const int l16  = lane & 15;

    const int m0 = (blockIdx.y * 8 + wave) * 16;
    const int n0 = blockIdx.x * (16 * NT);

    // A: lanes 16-31 carry K 16..31 ; B: lanes 16-31 carry K 16..31
    const bf16* ap = A + (long)(m0 + l16) * lda + (half ? 16 : 0);
    const bf16* bp[NT];
    #pragma unroll
    for (int t = 0; t < NT; ++t)
        bp[t] = W + (long)(n0 + t * 16 + l16) * ldb + (half ? 16 : 0);

    v8f acc[NT];
    v8f zero = {};
    #pragma unroll
    for (int t = 0; t < NT; ++t) acc[t] = zero;

    for (int k0 = 0; k0 < K; k0 += 32) {
        v16bf afrag = ld_frag(ap, ap + 8);
        ap += 32;
        #pragma unroll
        for (int t = 0; t < NT; ++t) {
            v16bf bfrag = ld_frag(bp[t], bp[t] + 8);
            bp[t] += 32;
            acc[t] = wmma_bf16(afrag, bfrag, acc[t]);
        }
    }

    const long rbase = (long)(m0 + half * 8) * ldc + n0 + l16;
    if (accum) {
        #pragma unroll
        for (int t = 0; t < NT; ++t)
            #pragma unroll
            for (int r = 0; r < 8; ++r)
                C[rbase + (long)r * ldc + t * 16] += scale * acc[t][r];
    } else {
        #pragma unroll
        for (int t = 0; t < NT; ++t)
            #pragma unroll
            for (int r = 0; r < 8; ++r)
                C[rbase + (long)r * ldc + t * 16] = scale * acc[t][r];
    }
}

// ---------------------------------------------------------------------------
// LayerNorm (over D_MODEL) + cast to bf16.  One block (256 thr) per token.
// ---------------------------------------------------------------------------
static __device__ float tdb_block_sum(float v) {
    __shared__ float red[9];
    __syncthreads();
    #pragma unroll
    for (int d = 16; d > 0; d >>= 1) v += __shfl_xor(v, d);
    if ((threadIdx.x & 31) == 0) red[threadIdx.x >> 5] = v;
    __syncthreads();
    if (threadIdx.x == 0) {
        float s = 0.f;
        #pragma unroll
        for (int i = 0; i < 8; ++i) s += red[i];
        red[8] = s;
    }
    __syncthreads();
    return red[8];
}

__global__ void tdb_layernorm_bf16(const float* __restrict__ x,
                                   const float* __restrict__ g,
                                   const float* __restrict__ b,
                                   bf16* __restrict__ out) {
    const long t = blockIdx.x;
    const float* xp = x + t * D_MODEL;
    float vals[8];
    float s = 0.f;
    #pragma unroll
    for (int i = 0; i < 8; ++i) {
        vals[i] = xp[threadIdx.x + i * 256];
        s += vals[i];
    }
    const float mu = tdb_block_sum(s) * (1.0f / D_MODEL);
    float s2 = 0.f;
    #pragma unroll
    for (int i = 0; i < 8; ++i) {
        float d = vals[i] - mu;
        s2 += d * d;
    }
    const float var = tdb_block_sum(s2) * (1.0f / D_MODEL);
    const float rstd = rsqrtf(var + 1e-6f);
    #pragma unroll
    for (int i = 0; i < 8; ++i) {
        int j = threadIdx.x + i * 256;
        out[t * D_MODEL + j] = (bf16)((vals[i] - mu) * rstd * g[j] + b[j]);
    }
}

// ---------------------------------------------------------------------------
// fp32 -> bf16 converters (flat and transposing, for LoRA A/B weights)
// ---------------------------------------------------------------------------
__global__ void tdb_cvt_bf16(const float* __restrict__ in, bf16* __restrict__ out, long n) {
    long i = (long)blockIdx.x * 256 + threadIdx.x;
    if (i < n) out[i] = (bf16)in[i];
}

__global__ void tdb_cvt_bf16_t(const float* __restrict__ in, bf16* __restrict__ out,
                               int rows, int cols) {
    long i = (long)blockIdx.x * 256 + threadIdx.x;
    if (i < (long)rows * cols) {
        int r = (int)(i / cols);
        int c = (int)(i - (long)r * cols);
        out[(long)c * rows + r] = (bf16)in[i];
    }
}

// ---------------------------------------------------------------------------
// Split fused qkv [T, 3*D_MODEL] into q/k (bf16 [bh, s, d], q pre-scaled by
// 1/sqrt(hd)) and vT (bf16 [bh, d, s]) for K-contiguous WMMA B-fragments.
// ---------------------------------------------------------------------------
__global__ void tdb_split_qkv(const float* __restrict__ qkv,
                              bf16* __restrict__ q, bf16* __restrict__ k,
                              bf16* __restrict__ vt) {
    long i = (long)blockIdx.x * 256 + threadIdx.x;   // over T * D_MODEL
    if (i >= (long)TOKENS * D_MODEL) return;
    const int t = (int)(i >> 11);          // token
    const int c = (int)(i & 2047);         // h*128 + d
    const int h = c >> 7;
    const int d = c & 127;
    const int b = t >> 11;                 // SEQ == 2048
    const int s = t & 2047;
    const int bh = b * N_HEADS + h;
    const float* src = qkv + (long)t * (3 * D_MODEL);
    const long qi = ((long)bh * SEQ + s) * HEAD_DIM + d;
    q[qi]  = (bf16)(src[c] * 0.08838834764831845f);            // 1/sqrt(128)
    k[qi]  = (bf16)(src[D_MODEL + c]);
    vt[((long)bh * HEAD_DIM + d) * SEQ + s] = (bf16)(src[2 * D_MODEL + c]);
}

// ---------------------------------------------------------------------------
// Fused flash-style attention with WMMA.
// Block = 8 waves; each wave owns one 16-row query tile of one (b,h).
// Q tile is staged once into LDS (4 KB/wave) and fragments are re-read per
// k-step, so no 32-VGPR Q set stays live across the 64-chunk loop (this was
// the spill source). Alibi values for a chunk are prefetched into registers
// before the WMMA chain so their loads overlap matrix math.
// ---------------------------------------------------------------------------
__global__ void __launch_bounds__(256, 1)
tdb_attention(const bf16* __restrict__ q,
              const bf16* __restrict__ k,
              const bf16* __restrict__ vt,
              const float* __restrict__ alibi,
              bf16* __restrict__ o) {
    __shared__ __align__(16) bf16 qstage[8][16][HEAD_DIM];   // 32 KB
    __shared__ __align__(16) bf16 pstage[8][16][32];         //  8 KB

    const int lane = threadIdx.x & 31;
    const int wave = threadIdx.x >> 5;
    const int half = lane >> 4;
    const int l16  = lane & 15;

    const int blocksPerBH = (SEQ / 16) / 8;          // 16
    const int bh = blockIdx.x / blocksPerBH;
    const int qg = blockIdx.x % blocksPerBH;
    const int q0 = (qg * 8 + wave) * 16;
    const int h  = bh & 15;

    // ---- stage this wave's 16x128 Q tile into LDS ----
    {
        const bf16* src = q + ((long)bh * SEQ + q0) * HEAD_DIM;
        bf16* dst = &qstage[wave][0][0];
        #pragma unroll
        for (int i = 0; i < 8; ++i) {
            const int idx = (i * 32 + lane) * 8;     // 2048 bf16 per tile
            *(v8bf*)(dst + idx) = *(const v8bf*)(src + idx);
        }
    }
    __syncthreads();

    // Loop pointers (advanced per 32-key chunk)
    const bf16*  kp = k  + (long)bh * SEQ * HEAD_DIM
                         + (long)l16 * HEAD_DIM + (half ? 16 : 0);
    const bf16*  vp = vt + (long)bh * HEAD_DIM * SEQ
                         + (long)l16 * SEQ + (half ? 16 : 0);
    const float* abp = alibi + (long)h * SEQ * SEQ
                             + (long)(q0 + half * 8) * SEQ + l16;
    const bf16* qp = &qstage[wave][l16][half ? 16 : 0];   // A-fragment base

    v8f o_acc[8];
    v8f zero = {};
    #pragma unroll
    for (int nt = 0; nt < 8; ++nt) o_acc[nt] = zero;
    float mrow[8], lrow[8];
    #pragma unroll
    for (int r = 0; r < 8; ++r) { mrow[r] = -3.0e38f; lrow[r] = 0.f; }

    for (int s0 = 0; s0 < SEQ; s0 += 32) {
        // ---- prefetch alibi for this chunk (overlaps the WMMA chain) ----
        float alb[2][8];
        #pragma unroll
        for (int t = 0; t < 2; ++t)
            #pragma unroll
            for (int r = 0; r < 8; ++r)
                alb[t][r] = abp[(long)r * SEQ + t * 16];

        // ---- scores: two 16x16 tiles over this 32-key chunk ----
        v8f st[2];
        st[0] = zero; st[1] = zero;
        #pragma unroll
        for (int kk = 0; kk < 4; ++kk) {
            v16bf aq = ld_frag(qp + kk * 32, qp + kk * 32 + 8);
            #pragma unroll
            for (int t = 0; t < 2; ++t) {
                const bf16* bp = kp + t * (16 * HEAD_DIM) + kk * 32;
                v16bf bfrag = ld_frag(bp, bp + 8);
                st[t] = wmma_bf16(aq, bfrag, st[t]);
            }
        }
        #pragma unroll
        for (int t = 0; t < 2; ++t)
            #pragma unroll
            for (int r = 0; r < 8; ++r)
                st[t][r] += alb[t][r];

        // ---- online softmax update ----
        #pragma unroll
        for (int r = 0; r < 8; ++r) {
            float mc = fmaxf(st[0][r], st[1][r]);
            mc = fmaxf(mc, __shfl_xor(mc, 1));
            mc = fmaxf(mc, __shfl_xor(mc, 2));
            mc = fmaxf(mc, __shfl_xor(mc, 4));
            mc = fmaxf(mc, __shfl_xor(mc, 8));
            const float mnew = fmaxf(mrow[r], mc);
            const float corr = __expf(mrow[r] - mnew);
            mrow[r] = mnew;
            const float p0 = __expf(st[0][r] - mnew);
            const float p1 = __expf(st[1][r] - mnew);
            st[0][r] = p0; st[1][r] = p1;
            float rs = p0 + p1;
            rs += __shfl_xor(rs, 1);
            rs += __shfl_xor(rs, 2);
            rs += __shfl_xor(rs, 4);
            rs += __shfl_xor(rs, 8);
            lrow[r] = lrow[r] * corr + rs;
            #pragma unroll
            for (int nt = 0; nt < 8; ++nt) o_acc[nt][r] *= corr;
        }

        // ---- stage P (C/D layout) into LDS, reload as A fragment ----
        #pragma unroll
        for (int t = 0; t < 2; ++t)
            #pragma unroll
            for (int r = 0; r < 8; ++r)
                pstage[wave][r + half * 8][t * 16 + l16] = (bf16)st[t][r];
        __syncthreads();
        const bf16* pp = &pstage[wave][l16][half ? 16 : 0];
        v16bf aP = ld_frag(pp, pp + 8);
        __syncthreads();

        // ---- O += P * V  (V transposed: column d contiguous along s) ----
        #pragma unroll
        for (int nt = 0; nt < 8; ++nt) {
            const bf16* bp = vp + (long)nt * (16 * SEQ);
            v16bf bfrag = ld_frag(bp, bp + 8);
            o_acc[nt] = wmma_bf16(aP, bfrag, o_acc[nt]);
        }
        kp  += 32 * HEAD_DIM;
        vp  += 32;
        abp += 32;
    }

    // ---- normalize and store o in [b, s, h*128+d] (bf16) ----
    bf16* op = o + ((long)(bh >> 4) * SEQ + q0 + half * 8) * D_MODEL
                 + h * HEAD_DIM + l16;
    #pragma unroll
    for (int r = 0; r < 8; ++r) {
        const float inv = 1.0f / lrow[r];
        #pragma unroll
        for (int nt = 0; nt < 8; ++nt)
            op[(long)r * D_MODEL + nt * 16] = (bf16)(o_acc[nt][r] * inv);
    }
}

// ---------------------------------------------------------------------------
// Elementwise: residual + layer-scale ; SwiGLU gate
// ---------------------------------------------------------------------------
__global__ void tdb_residual_ls(const float* __restrict__ base,
                                const float* __restrict__ acc,
                                const float* __restrict__ gamma,
                                float* __restrict__ out, long n) {
    long i = (long)blockIdx.x * 256 + threadIdx.x;
    if (i < n) out[i] = base[i] + gamma[i & (D_MODEL - 1)] * acc[i];
}

__global__ void tdb_silu_gate(const float* __restrict__ h, bf16* __restrict__ g) {
    long i = (long)blockIdx.x * 256 + threadIdx.x;   // over T * D_FF
    if (i >= (long)TOKENS * D_FF) return;
    const long t = i / D_FF;
    const long j = i - t * D_FF;
    const float gate = h[t * (2 * D_FF) + j];
    const float act  = h[t * (2 * D_FF) + D_FF + j];
    const float s = gate / (1.0f + __expf(-gate));
    g[i] = (bf16)(s * act);
}

// ---------------------------------------------------------------------------
// Host-side orchestration
// ---------------------------------------------------------------------------
template <int NT>
static void launch_gemm(const void* A, int lda, const void* W, int ldb,
                        float* C, int ldc, int N, int K, float scale, int accum,
                        hipStream_t stream) {
    dim3 grid(N / (16 * NT), TOKENS / 128);   // M = 4096 always
    tdb_gemm_bf16<NT><<<grid, 256, 0, stream>>>((const bf16*)A, lda,
                                                (const bf16*)W, ldb,
                                                C, ldc, K, scale, accum);
}

static void launch_cvt(const float* in, void* out, long n, hipStream_t s) {
    tdb_cvt_bf16<<<(unsigned)((n + 255) / 256), 256, 0, s>>>(in, (bf16*)out, n);
}
static void launch_cvt_t(const float* in, void* out, int rows, int cols, hipStream_t s) {
    long n = (long)rows * cols;
    tdb_cvt_bf16_t<<<(unsigned)((n + 255) / 256), 256, 0, s>>>(in, (bf16*)out, rows, cols);
}

extern "C" void kernel_launch(void* const* d_in, const int* in_sizes, int n_in,
                              void* d_out, int out_size, void* d_ws, size_t ws_size,
                              hipStream_t stream) {
    (void)in_sizes; (void)n_in; (void)out_size; (void)ws_size;

    const float* x      = (const float*)d_in[0];
    const float* alibi  = (const float*)d_in[1];
    const float* ln1_g  = (const float*)d_in[2];
    const float* ln1_b  = (const float*)d_in[3];
    const float* ln2_g  = (const float*)d_in[4];
    const float* ln2_b  = (const float*)d_in[5];
    const float* qkv_W  = (const float*)d_in[6];
    const float* qkv_A  = (const float*)d_in[7];
    const float* qkv_B  = (const float*)d_in[8];
    const float* out_W  = (const float*)d_in[9];
    const float* out_A  = (const float*)d_in[10];
    const float* out_B  = (const float*)d_in[11];
    const float* fc1_W  = (const float*)d_in[12];
    const float* fc1_A  = (const float*)d_in[13];
    const float* fc1_B  = (const float*)d_in[14];
    const float* fc2_W  = (const float*)d_in[15];
    const float* fc2_A  = (const float*)d_in[16];
    const float* fc2_B  = (const float*)d_in[17];
    const float* ls1    = (const float*)d_in[18];
    const float* ls2    = (const float*)d_in[19];
    float* outp = (float*)d_out;

    // ---- workspace arena (WBUF reused for every converted weight) ----
    char* w = (char*)d_ws;
    size_t off = 0;
    auto carve = [&](size_t bytes) { void* p = w + off; off += (bytes + 255) & ~255ull; return p; };
    void* WBUF = carve((size_t)2 * D_FF * D_MODEL * 2);          // 64 MiB (max weight bf16)
    void* ATB  = carve((size_t)RANK * D_FF * 2);                 // LoRA A^T bf16
    void* BTB  = carve((size_t)2 * D_FF * RANK * 2);             // LoRA B^T bf16
    void* XN   = carve((size_t)TOKENS * D_MODEL * 2);            // layernormed x, bf16
    float* BIG = (float*)carve((size_t)TOKENS * 2 * D_FF * 4);   // qkv out, then h
    float* XA  = (float*)carve((size_t)TOKENS * RANK * 4);       // LoRA mid, f32
    void* XAB  = carve((size_t)TOKENS * RANK * 2);               // LoRA mid, bf16
    void* QB   = carve((size_t)TOKENS * D_MODEL * 2);
    void* KB   = carve((size_t)TOKENS * D_MODEL * 2);
    void* VTB  = carve((size_t)TOKENS * D_MODEL * 2);
    void* OB   = carve((size_t)TOKENS * D_MODEL * 2);
    float* ACC = (float*)carve((size_t)TOKENS * D_MODEL * 4);
    float* X2  = (float*)carve((size_t)TOKENS * D_MODEL * 4);
    void* GB   = carve((size_t)TOKENS * D_FF * 2);

    const long nTD = (long)TOKENS * D_MODEL;                     // 8388608
    const float lora_s = 1.0f / RANK;

    // ===== attention branch =====
    tdb_layernorm_bf16<<<TOKENS, 256, 0, stream>>>(x, ln1_g, ln1_b, (bf16*)XN);

    launch_cvt(qkv_W, WBUF, (long)3 * D_MODEL * D_MODEL, stream);
    launch_gemm<4>(XN, D_MODEL, WBUF, D_MODEL, BIG, 3 * D_MODEL,
                   3 * D_MODEL, D_MODEL, 1.0f, 0, stream);
    launch_cvt_t(qkv_A, ATB, D_MODEL, RANK, stream);
    launch_gemm<2>(XN, D_MODEL, ATB, D_MODEL, XA, RANK, RANK, D_MODEL, 1.0f, 0, stream);
    launch_cvt(XA, XAB, (long)TOKENS * RANK, stream);
    launch_cvt_t(qkv_B, BTB, RANK, 3 * D_MODEL, stream);
    launch_gemm<4>(XAB, RANK, BTB, RANK, BIG, 3 * D_MODEL,
                   3 * D_MODEL, RANK, lora_s, 1, stream);

    tdb_split_qkv<<<(unsigned)(nTD / 256), 256, 0, stream>>>(
        BIG, (bf16*)QB, (bf16*)KB, (bf16*)VTB);

    tdb_attention<<<BATCH * N_HEADS * (SEQ / 16) / 8, 256, 0, stream>>>(
        (const bf16*)QB, (const bf16*)KB, (const bf16*)VTB, alibi, (bf16*)OB);

    launch_cvt(out_W, WBUF, (long)D_MODEL * D_MODEL, stream);
    launch_gemm<4>(OB, D_MODEL, WBUF, D_MODEL, ACC, D_MODEL,
                   D_MODEL, D_MODEL, 1.0f, 0, stream);
    launch_cvt_t(out_A, ATB, D_MODEL, RANK, stream);
    launch_gemm<2>(OB, D_MODEL, ATB, D_MODEL, XA, RANK, RANK, D_MODEL, 1.0f, 0, stream);
    launch_cvt(XA, XAB, (long)TOKENS * RANK, stream);
    launch_cvt_t(out_B, BTB, RANK, D_MODEL, stream);
    launch_gemm<4>(XAB, RANK, BTB, RANK, ACC, D_MODEL, D_MODEL, RANK, lora_s, 1, stream);

    tdb_residual_ls<<<(unsigned)(nTD / 256), 256, 0, stream>>>(x, ACC, ls1, X2, nTD);

    // ===== MLP branch =====
    tdb_layernorm_bf16<<<TOKENS, 256, 0, stream>>>(X2, ln2_g, ln2_b, (bf16*)XN);

    launch_cvt(fc1_W, WBUF, (long)2 * D_FF * D_MODEL, stream);
    launch_gemm<4>(XN, D_MODEL, WBUF, D_MODEL, BIG, 2 * D_FF,
                   2 * D_FF, D_MODEL, 1.0f, 0, stream);
    launch_cvt_t(fc1_A, ATB, D_MODEL, RANK, stream);
    launch_gemm<2>(XN, D_MODEL, ATB, D_MODEL, XA, RANK, RANK, D_MODEL, 1.0f, 0, stream);
    launch_cvt(XA, XAB, (long)TOKENS * RANK, stream);
    launch_cvt_t(fc1_B, BTB, RANK, 2 * D_FF, stream);
    launch_gemm<4>(XAB, RANK, BTB, RANK, BIG, 2 * D_FF, 2 * D_FF, RANK, lora_s, 1, stream);

    tdb_silu_gate<<<(unsigned)(((long)TOKENS * D_FF) / 256), 256, 0, stream>>>(
        BIG, (bf16*)GB);

    launch_cvt(fc2_W, WBUF, (long)D_MODEL * D_FF, stream);
    launch_gemm<4>(GB, D_FF, WBUF, D_FF, ACC, D_MODEL, D_MODEL, D_FF, 1.0f, 0, stream);
    launch_cvt_t(fc2_A, ATB, D_FF, RANK, stream);
    launch_gemm<2>(GB, D_FF, ATB, D_FF, XA, RANK, RANK, D_FF, 1.0f, 0, stream);
    launch_cvt(XA, XAB, (long)TOKENS * RANK, stream);
    launch_cvt_t(fc2_B, BTB, RANK, D_MODEL, stream);
    launch_gemm<4>(XAB, RANK, BTB, RANK, ACC, D_MODEL, D_MODEL, RANK, lora_s, 1, stream);

    tdb_residual_ls<<<(unsigned)(nTD / 256), 256, 0, stream>>>(X2, ACC, ls2, outp, nTD);
}
